// STTLoraLinear_33844342292634
// MI455X (gfx1250) — compile-verified, matrix operand order checked by reading, not run on hardware
//
#include <hip/hip_runtime.h>

// ---------------------------------------------------------------------------
// STTLoraLinear for MI455X (gfx1250, wave32, WMMA + async LDS DMA).
//
// out = scatter_cols( x[:, in_idx] @ W_eff^T + b, out_idx ),
//   W_eff = W_base + 2 * lora_B @ lora_A      (LoRA folded into the weights)
//
// Precision: split-bf16 (hi+lo) with f32 WMMA accumulation:
//   x*w ~= xh*wh + xh*wl + xl*wh   (error ~2^-17, fp32-grade)
//
// Tiling chosen for LDS-read bandwidth (~256B/clk/WGP): wave tile 64x64
// gives 48 WMMAs per 32 ds_load_b128 per K-chunk (1.5 wmma/load), block
// tile 128(M) x 256(N) with 8 waves. Double-buffered LDS (96 KB) filled by
// GLOBAL_LOAD_ASYNC_TO_LDS_B128 (ASYNCcnt), one barrier per K-chunk.
// Packed operands (80 MB) stay resident in the 192 MB L2.
// ---------------------------------------------------------------------------

typedef __attribute__((ext_vector_type(16))) __bf16        v16bf;
typedef __attribute__((ext_vector_type(8)))  float         v8f;
typedef __attribute__((ext_vector_type(4)))  unsigned int  v4u;

union Frag { v16bf bf; v4u u[2]; };

#define ROWS  8192   // 2*4096 flattened batch rows
#define KDIM  2048   // ACTIVE (gathered inputs)
#define ODIM  2048   // ACTIVE outputs
#define OUTD  4096   // ORIG_OUT
#define INDIM 4096   // ORIG_IN

__device__ __forceinline__ unsigned short f2bf(float f) {
    unsigned u = __float_as_uint(f);
    unsigned r = ((u >> 16) & 1u) + 0x7FFFu;   // round-to-nearest-even
    return (unsigned short)((u + r) >> 16);
}
__device__ __forceinline__ float bf2f(unsigned short h) {
    return __uint_as_float(((unsigned)h) << 16);
}

// LDS byte offset of a shared-memory pointer (addrspace(3) ptrs are the raw
// offsets the async-DMA VDST operand wants).
__device__ __forceinline__ unsigned lds_off(const void* p) {
    return (unsigned)(unsigned long long)
        (const __attribute__((address_space(3))) void*)p;
}

// Two async 16B copies global->LDS; `offset:16` advances BOTH the LDS and
// global addresses per the ISA's async-load pseudocode, matching our layout.
__device__ __forceinline__ void async_cp32(unsigned ldsOff,
                                           const unsigned short* base,
                                           unsigned gByteOff) {
    asm volatile("global_load_async_to_lds_b128 %0, %1, %2\n\t"
                 "global_load_async_to_lds_b128 %0, %1, %2 offset:16"
                 :
                 : "v"(ldsOff), "v"(gByteOff), "s"(base)
                 : "memory");
}

__device__ __forceinline__ void wait_async_all() {
    asm volatile("s_wait_asynccnt 0x0" ::: "memory");
}

// ---- zero the 8192x4096 output ------------------------------------------
__global__ __launch_bounds__(256) void zero_out_k(float4* __restrict__ out) {
    size_t i = (size_t)blockIdx.x * 256 + threadIdx.x;
    out[i] = float4{0.f, 0.f, 0.f, 0.f};
}

// ---- gather x columns and split into planar bf16 hi/lo -------------------
__global__ __launch_bounds__(256) void pack_x_k(const float* __restrict__ x,
                                                const int* __restrict__ iidx,
                                                unsigned short* __restrict__ xhi,
                                                unsigned short* __restrict__ xlo) {
    size_t id = (size_t)blockIdx.x * 256 + threadIdx.x;   // ROWS*KDIM
    int n = (int)(id >> 11);
    int a = (int)(id & 2047);
    float v = x[(size_t)n * INDIM + iidx[a]];
    unsigned short h = f2bf(v);
    unsigned short l = f2bf(v - bf2f(h));
    xhi[id] = h;
    xlo[id] = l;
}

// ---- fold LoRA into W, split into planar bf16 hi/lo ----------------------
__global__ __launch_bounds__(256) void pack_w_k(const float* __restrict__ W,
                                                const float* __restrict__ lA,
                                                const float* __restrict__ lB,
                                                unsigned short* __restrict__ whi,
                                                unsigned short* __restrict__ wlo) {
    size_t id = (size_t)blockIdx.x * 256 + threadIdx.x;   // ODIM*KDIM
    int o = (int)(id >> 11);
    int a = (int)(id & 2047);
    float v = W[id];
    float s = 0.f;
    #pragma unroll
    for (int r = 0; r < 16; ++r)
        s += lB[o * 16 + r] * lA[(size_t)r * KDIM + a];
    v += 2.0f * s;                                         // SCALING = 32/16
    unsigned short h = f2bf(v);
    unsigned short l = f2bf(v - bf2f(h));
    whi[id] = h;
    wlo[id] = l;
}

// ---- main split-bf16 WMMA GEMM + bias + scatter --------------------------
// Block: 256 threads (8 waves), tile 128(M) x 256(N), K-chunk 32,
// double-buffered LDS filled by async DMA.
// Wave (wm,wn): wm in {0,1} -> 64 rows, wn in {0..3} -> 64 cols.
__global__ __launch_bounds__(256) void stt_gemm_k(const unsigned short* __restrict__ xhi,
                                                  const unsigned short* __restrict__ xlo,
                                                  const unsigned short* __restrict__ whi,
                                                  const unsigned short* __restrict__ wlo,
                                                  const float* __restrict__ bias,
                                                  const int* __restrict__ oidx,
                                                  float* __restrict__ out) {
    // A: [buf][hi|lo][128][32]  2*2*128*32*2B = 32 KB
    // B: [buf][hi|lo][256][32]  2*2*256*32*2B = 64 KB   => 96 KB total
    __shared__ __align__(16) unsigned short sA[2][2][128][32];
    __shared__ __align__(16) unsigned short sB[2][2][256][32];

    const int tid   = threadIdx.x;
    const int wave  = tid >> 5;
    const int lane  = tid & 31;
    const int wm    = wave & 1;
    const int wn    = wave >> 1;
    const int lrow  = lane & 15;   // M row / output col within 16x16 tile
    const int lhalf = lane >> 4;   // K sub-chunk select per ISA 16-bit layout

    const int bm = blockIdx.y;     // 0..63  (M blocks, 128 rows each)
    const int bn = blockIdx.x;     // 0..7   (N blocks, 256 cols each)

    // DMA-fill mapping: A -> thread covers 32B of one row-half;
    //                   B -> thread covers one full 64B row.
    const int ldrA = tid >> 1;           // 0..127
    const int ldcA = (tid & 1) << 4;     // 0 or 16 (elements)

    const unsigned gA0 = ((unsigned)(bm * 128 + ldrA) * KDIM + ldcA) * 2u;
    const unsigned gB0 = ((unsigned)(bn * 256 + tid) * KDIM) * 2u;

    unsigned lAh[2], lAl[2], lBh[2], lBl[2];
    #pragma unroll
    for (int b = 0; b < 2; ++b) {
        lAh[b] = lds_off(&sA[b][0][ldrA][ldcA]);
        lAl[b] = lds_off(&sA[b][1][ldrA][ldcA]);
        lBh[b] = lds_off(&sB[b][0][tid][0]);
        lBl[b] = lds_off(&sB[b][1][tid][0]);
    }

    v8f acc[4][4];
    #pragma unroll
    for (int i = 0; i < 4; ++i)
        #pragma unroll
        for (int j = 0; j < 4; ++j)
            acc[i][j] = (v8f){0.f, 0.f, 0.f, 0.f, 0.f, 0.f, 0.f, 0.f};

    // Prologue: start DMA for chunk 0 into buffer 0.
    async_cp32(lAh[0], xhi, gA0);
    async_cp32(lAl[0], xlo, gA0);
    async_cp32(lBh[0],      whi, gB0);
    async_cp32(lBh[0] + 32, whi, gB0 + 32);
    async_cp32(lBl[0],      wlo, gB0);
    async_cp32(lBl[0] + 32, wlo, gB0 + 32);

    int buf = 0;
    for (int kc = 0; kc < KDIM; kc += 32) {
        // Publish buffer `buf`; after this barrier every wave has also
        // finished computing on buf^1, so refilling it below is safe.
        wait_async_all();
        __syncthreads();

        if (kc + 32 < KDIM) {
            const unsigned dk = (unsigned)(kc + 32) * 2u;
            const int nb = buf ^ 1;
            async_cp32(lAh[nb], xhi, gA0 + dk);
            async_cp32(lAl[nb], xlo, gA0 + dk);
            async_cp32(lBh[nb],      whi, gB0 + dk);
            async_cp32(lBh[nb] + 32, whi, gB0 + dk + 32);
            async_cp32(lBl[nb],      wlo, gB0 + dk);
            async_cp32(lBl[nb] + 32, wlo, gB0 + dk + 32);
        }

        // Fragment lane mapping per the ISA 16-bit operand layout:
        // lanes 0-15 -> K{0..7,16..23}, lanes 16-31 -> K{8..15,24..31}.
        // Identical mapping for A and B, so any consistent K permutation
        // cancels in the dot product.
        Frag Bh[4], Bl[4];
        #pragma unroll
        for (int tn = 0; tn < 4; ++tn) {
            const int c = wn * 64 + tn * 16 + lrow;
            const unsigned short* p = &sB[buf][0][c][lhalf * 8];
            Bh[tn].u[0] = *(const v4u*)p;
            Bh[tn].u[1] = *(const v4u*)(p + 16);
            const unsigned short* q = &sB[buf][1][c][lhalf * 8];
            Bl[tn].u[0] = *(const v4u*)q;
            Bl[tn].u[1] = *(const v4u*)(q + 16);
        }

        #pragma unroll
        for (int tm = 0; tm < 4; ++tm) {
            const int r = wm * 64 + tm * 16 + lrow;
            Frag Ah, Al;
            const unsigned short* p = &sA[buf][0][r][lhalf * 8];
            Ah.u[0] = *(const v4u*)p;
            Ah.u[1] = *(const v4u*)(p + 16);
            const unsigned short* q = &sA[buf][1][r][lhalf * 8];
            Al.u[0] = *(const v4u*)q;
            Al.u[1] = *(const v4u*)(q + 16);
            #pragma unroll
            for (int tn = 0; tn < 4; ++tn) {
                acc[tm][tn] = __builtin_amdgcn_wmma_f32_16x16x32_bf16(
                    false, Ah.bf, false, Bh[tn].bf, (short)0, acc[tm][tn], false, false);
                acc[tm][tn] = __builtin_amdgcn_wmma_f32_16x16x32_bf16(
                    false, Ah.bf, false, Bl[tn].bf, (short)0, acc[tm][tn], false, false);
                acc[tm][tn] = __builtin_amdgcn_wmma_f32_16x16x32_bf16(
                    false, Al.bf, false, Bh[tn].bf, (short)0, acc[tm][tn], false, false);
            }
        }
        buf ^= 1;
    }

    // Epilogue: C/D layout -> lane L: N = L%16, M = vgpr + (L>=16 ? 8 : 0).
    #pragma unroll
    for (int tm = 0; tm < 4; ++tm) {
        const int n0 = bm * 128 + wm * 64 + tm * 16 + lhalf * 8;
        #pragma unroll
        for (int tn = 0; tn < 4; ++tn) {
            const int o  = bn * 256 + wn * 64 + tn * 16 + lrow;
            const float bs = bias[o];
            const int   oc = oidx[o];
            #pragma unroll
            for (int i = 0; i < 8; ++i)
                out[(size_t)(n0 + i) * OUTD + oc] = acc[tm][tn][i] + bs;
        }
    }
}

// ---------------------------------------------------------------------------
extern "C" void kernel_launch(void* const* d_in, const int* in_sizes, int n_in,
                              void* d_out, int out_size, void* d_ws, size_t ws_size,
                              hipStream_t stream) {
    const float* x    = (const float*)d_in[0];
    const float* W    = (const float*)d_in[1];
    const float* b    = (const float*)d_in[2];
    const float* lA   = (const float*)d_in[3];
    const float* lB   = (const float*)d_in[4];
    const int*   iidx = (const int*)d_in[5];
    const int*   oidx = (const int*)d_in[6];
    float*       out  = (float*)d_out;

    // Workspace layout (80 MB total):
    //   xhi 32MB | xlo 32MB | whi 8MB | wlo 8MB
    char* ws = (char*)d_ws;
    unsigned short* xhi = (unsigned short*)(ws);
    unsigned short* xlo = (unsigned short*)(ws + (size_t)ROWS * KDIM * 2);
    unsigned short* whi = (unsigned short*)(ws + (size_t)ROWS * KDIM * 4);
    unsigned short* wlo = (unsigned short*)(ws + (size_t)ROWS * KDIM * 4
                                               + (size_t)ODIM * KDIM * 2);

    zero_out_k<<<(ROWS * (size_t)OUTD / 4) / 256, 256, 0, stream>>>((float4*)out);
    pack_x_k<<<(ROWS * (size_t)KDIM) / 256, 256, 0, stream>>>(x, iidx, xhi, xlo);
    pack_w_k<<<(ODIM * (size_t)KDIM) / 256, 256, 0, stream>>>(W, lA, lB, whi, wlo);

    dim3 grid(ODIM / 256, ROWS / 128);   // (8, 64)
    stt_gemm_k<<<grid, 256, 0, stream>>>(xhi, xlo, whi, wlo, b, oidx, out);
}
